// HyperbolicDistanceHead_29188597743690
// MI455X (gfx1250) — matrix operand order repaired and checked by reading, not compiled, not run on hardware
//
#include <hip/hip_runtime.h>

#define TOK 2048
#define VOC 32000
#define DIM 768
#define KT 32
#define KITERS (DIM / KT)   // 24
#define MBLK 128
#define NBLK 128
#define LDSS 40             // padded LDS row stride in elements (80B row: 64B data + 16B pad)
#define EPSF 1e-5f

typedef __attribute__((ext_vector_type(8)))  __bf16 bf16x8;
typedef __attribute__((ext_vector_type(16))) __bf16 bf16x16;
typedef __attribute__((ext_vector_type(8)))  float  floatx8;
typedef __attribute__((ext_vector_type(4)))  unsigned int u32x4;
typedef __attribute__((ext_vector_type(8)))  int i32x8;
typedef __attribute__((ext_vector_type(4)))  int i32x4;

// ---------------------------------------------------------------------------
// Kernel 1: Poincare-ball projection + squared norms + fp32 -> bf16 convert.
// One block (256 threads = 8 waves) per row of 768 elements.
// ---------------------------------------------------------------------------
__global__ __launch_bounds__(256) void hyp_proj(
    const float* __restrict__ hs, const float* __restrict__ wt,
    __bf16* __restrict__ xbf, __bf16* __restrict__ ybf,
    float* __restrict__ xsq, float* __restrict__ ysq)
{
  const int row = blockIdx.x;
  const int t = threadIdx.x;
  const float* src;
  __bf16* dst;
  float* sq;
  int r;
  if (row < TOK) {
    r = row;       src = hs + (size_t)r * DIM; dst = xbf + (size_t)r * DIM; sq = xsq;
  } else {
    r = row - TOK; src = wt + (size_t)r * DIM; dst = ybf + (size_t)r * DIM; sq = ysq;
  }

  float v0 = src[t], v1 = src[t + 256], v2 = src[t + 512];
  float p = v0 * v0 + v1 * v1 + v2 * v2;
  #pragma unroll
  for (int off = 16; off > 0; off >>= 1) p += __shfl_xor(p, off, 32);

  __shared__ float red[8];
  __shared__ float s_scale;
  const int wave = t >> 5, lane = t & 31;
  if (lane == 0) red[wave] = p;
  __syncthreads();
  if (t == 0) {
    float s = 0.f;
    #pragma unroll
    for (int i = 0; i < 8; i++) s += red[i];
    float scale = fminf((1.0f - EPSF) / (sqrtf(s) + 1e-10f), 1.0f);
    s_scale = scale;
    sq[r] = scale * scale * s;   // ||proj(v)||^2
  }
  __syncthreads();
  const float scale = s_scale;
  dst[t]       = (__bf16)(v0 * scale);
  dst[t + 256] = (__bf16)(v1 * scale);
  dst[t + 512] = (__bf16)(v2 * scale);
}

// ---------------------------------------------------------------------------
// TDM: issue a 2D tile load (tile_dim0 = KT elems per row, tile_dim1 = 128
// rows, element = 2B bf16) from a row-major tensor with row stride DIM into
// LDS at lds_addr, inserting 16B of LDS padding after every 64B of data
// (-> 80B LDS row stride == LDSS elements).  D# per CDNA5 ISA ch.8.
// ---------------------------------------------------------------------------
__device__ __forceinline__ void tdm_load_tile(unsigned lds_addr,
                                              const __bf16* gptr,
                                              unsigned tensor_rows)
{
  const unsigned long long ga = (unsigned long long)(uintptr_t)gptr;
  u32x4 g0;
  g0[0] = 1u;                                            // count=1, user descriptor
  g0[1] = lds_addr;                                      // [63:32]   lds_addr
  g0[2] = (unsigned)ga;                                  // [95:64]   global_addr[31:0]
  g0[3] = (unsigned)((ga >> 32) & 0x01ffffffu)           // [120:96]  global_addr[56:32]
          | (2u << 30);                                  // [127:126] type=2 (image)
  i32x8 g1;
  g1[0] = (int)((1u << 16)      // data_size = 2 bytes
              | (1u << 20)      // pad_enable
              | (3u << 22)      // pad_interval: 16 DWORDs (64B) of data ...
              | (3u << 25));    // pad_amount:   4 DWORDs (16B) of pad
  g1[1] = (int)((DIM & 0xffffu) << 16);                  // tensor_dim0[15:0]
  g1[2] = (int)((DIM >> 16) | ((tensor_rows & 0xffffu) << 16));  // dim0 hi | dim1 lo
  g1[3] = (int)((tensor_rows >> 16) | ((unsigned)KT << 16));     // dim1 hi | tile_dim0
  g1[4] = (int)MBLK;                                     // tile_dim1 = 128 rows (tile_dim2=0)
  g1[5] = (int)DIM;                                      // tensor_dim0_stride[31:0]
  g1[6] = 0;                                             // stride hi | dim1_stride lo
  g1[7] = 0;
  const i32x4 z4 = {0, 0, 0, 0};                         // groups 2/3 unused (2D tensor)
  const i32x8 z8 = {0, 0, 0, 0, 0, 0, 0, 0};
  __builtin_amdgcn_tensor_load_to_lds(g0, g1, z4, z4, z8, 0);
}

// ---------------------------------------------------------------------------
// Kernel 2: fused  cross = X * Y^T  (bf16 WMMA, f32 accum)  +  arccosh epilogue.
// Block tile 128x128, 8 waves, each wave computes 32(M) x 64(N) = 8 WMMA tiles.
// Tiles fed by the Tensor Data Mover (wave0 -> A, wave1 -> B), double-buffered
// LDS, TENSORcnt + one barrier per K step.
// ---------------------------------------------------------------------------
__global__ __launch_bounds__(256) void hyp_gemm(
    const __bf16* __restrict__ X,    // [TOK][DIM] projected bf16
    const __bf16* __restrict__ Y,    // [VOC][DIM] projected bf16
    const float* __restrict__ xsq,   // [TOK]
    const float* __restrict__ ysq,   // [VOC]
    float* __restrict__ out)         // [TOK][VOC]
{
  __shared__ __bf16 As[2][MBLK * LDSS];
  __shared__ __bf16 Bs[2][NBLK * LDSS];

  const int t    = threadIdx.x;
  const int lane = t & 31;
  const int wave = t >> 5;
  const int mbase = blockIdx.y * MBLK;
  const int nbase = blockIdx.x * NBLK;
  const int wm = (wave & 3) * 32;   // wave M offset inside block tile
  const int wn = (wave >> 2) * 64;  // wave N offset inside block tile

  const __bf16* gA = X + (size_t)mbase * DIM;  // A tile origin (row, k=0)
  const __bf16* gB = Y + (size_t)nbase * DIM;  // B tile origin
  const unsigned ldsA[2] = { (unsigned)(uintptr_t)&As[0][0],
                             (unsigned)(uintptr_t)&As[1][0] };
  const unsigned ldsB[2] = { (unsigned)(uintptr_t)&Bs[0][0],
                             (unsigned)(uintptr_t)&Bs[1][0] };

  floatx8 acc[2][4];
  #pragma unroll
  for (int i = 0; i < 2; i++)
    #pragma unroll
    for (int j = 0; j < 4; j++) acc[i][j] = {};

  // Prime buffer 0 (k = 0..31): TDM DMA, one issuing wave per tile.
  if (wave == 0) {
    tdm_load_tile(ldsA[0], gA, MBLK);
    __builtin_amdgcn_s_wait_tensorcnt(0);
  } else if (wave == 1) {
    tdm_load_tile(ldsB[0], gB, NBLK);
    __builtin_amdgcn_s_wait_tensorcnt(0);
  }
  __syncthreads();

  // Fragment addressing per CDNA5 16-bit WMMA VGPR layouts (wave32):
  //  A 16x32: lanes 0-15 row=l, K {0..7,16..23};  lanes 16-31 row=l-16, K {8..15,24..31}
  //  B 32x16: lanes 0-15 col=l, K {0..7,16..23};  lanes 16-31 col=l-16, K {8..15,24..31}
  const int lrow  = lane & 15;
  const int akoff = (lane < 16) ? 0 : 8;
  const int bkoff = (lane < 16) ? 0 : 8;

  int buf = 0;
  for (int kk = 0; kk < KITERS; kk++) {
    // Kick off the DMA for the next K tile into the other buffer.
    if (kk + 1 < KITERS) {
      const int ko = (kk + 1) * KT;
      if (wave == 0)      tdm_load_tile(ldsA[buf ^ 1], gA + ko, MBLK);
      else if (wave == 1) tdm_load_tile(ldsB[buf ^ 1], gB + ko, NBLK);
    }

    bf16x16 afrag[2];
    #pragma unroll
    for (int i = 0; i < 2; i++) {
      const __bf16* base = &As[buf][(wm + i * 16 + lrow) * LDSS + akoff];
      bf16x8 lo = *(const bf16x8*)(base);
      bf16x8 hi = *(const bf16x8*)(base + 16);
      afrag[i] = __builtin_shufflevector(lo, hi, 0,1,2,3,4,5,6,7,8,9,10,11,12,13,14,15);
    }
    #pragma unroll
    for (int j = 0; j < 4; j++) {
      const __bf16* base = &Bs[buf][(wn + j * 16 + lrow) * LDSS + bkoff];
      bf16x8 lo = *(const bf16x8*)(base);
      bf16x8 hi = *(const bf16x8*)(base + 16);
      bf16x16 bfrag = __builtin_shufflevector(lo, hi, 0,1,2,3,4,5,6,7,8,9,10,11,12,13,14,15);
      #pragma unroll
      for (int i = 0; i < 2; i++) {
        acc[i][j] = __builtin_amdgcn_wmma_f32_16x16x32_bf16(
            false, afrag[i], false, bfrag, (short)0, acc[i][j], false, false);
      }
    }

    if (kk + 1 < KITERS) {
      if (wave <= 1) __builtin_amdgcn_s_wait_tensorcnt(0);  // DMA of next tile done
      __syncthreads();                                      // publish to all waves
    }
    buf ^= 1;
  }

  // Fused hyperbolic-distance epilogue.
  // C/D layout: lanes 0-15 -> M = vgpr_r, N = lane; lanes 16-31 -> M = vgpr_r+8, N = lane-16.
  const int halfoff = (lane < 16) ? 0 : 8;
  #pragma unroll
  for (int i = 0; i < 2; i++) {
    const int mrow = mbase + wm + i * 16 + halfoff;   // multiple of 8
    const float4 xa = *(const float4*)&xsq[mrow];
    const float4 xb = *(const float4*)&xsq[mrow + 4];
    const float xv[8] = {xa.x, xa.y, xa.z, xa.w, xb.x, xb.y, xb.z, xb.w};
    #pragma unroll
    for (int j = 0; j < 4; j++) {
      const int ncol = nbase + wn + j * 16 + lrow;
      const float ys = ysq[ncol];
      const float oneys = 1.0f - ys;
      #pragma unroll
      for (int r = 0; r < 8; r++) {
        const float xs    = xv[r];
        const float cross = acc[i][j][r];
        const float diff  = xs + ys - 2.0f * cross;           // ||x-y||^2
        const float denom = (1.0f - xs) * oneys;
        float arg = 1.0f + 2.0f * diff / (denom + EPSF);
        arg = fmaxf(arg, 1.0f + EPSF);
        const float dist = logf(arg + sqrtf(arg * arg - 1.0f));  // arccosh
        out[(size_t)(mrow + r) * VOC + ncol] = -dist;
      }
    }
  }
}

// ---------------------------------------------------------------------------
// Launcher
// ---------------------------------------------------------------------------
extern "C" void kernel_launch(void* const* d_in, const int* in_sizes, int n_in,
                              void* d_out, int out_size, void* d_ws, size_t ws_size,
                              hipStream_t stream) {
  const float* hs = (const float*)d_in[0];   // [2,1024,768] fp32
  const float* wt = (const float*)d_in[1];   // [32000,768] fp32
  float* out = (float*)d_out;                // [2,1024,32000] fp32

  char* ws = (char*)d_ws;
  const size_t off_xbf = 0;
  const size_t off_ybf = off_xbf + (size_t)TOK * DIM * 2;  //  3,145,728
  const size_t off_xsq = off_ybf + (size_t)VOC * DIM * 2;  // 52,297,728
  const size_t off_ysq = off_xsq + (size_t)TOK * 4;        // 52,305,920
  __bf16* xbf = (__bf16*)(ws + off_xbf);
  __bf16* ybf = (__bf16*)(ws + off_ybf);
  float*  xsq = (float*)(ws + off_xsq);
  float*  ysq = (float*)(ws + off_ysq);

  hyp_proj<<<TOK + VOC, 256, 0, stream>>>(hs, wt, xbf, ybf, xsq, ysq);

  dim3 grid(VOC / NBLK, TOK / MBLK);   // 250 x 16
  hyp_gemm<<<grid, 256, 0, stream>>>(xbf, ybf, xsq, ysq, out);
}
